// StaticPrunedMultiheadAttention_31482110279711
// MI455X (gfx1250) — matrix operand order repaired and checked by reading, not compiled, not run on hardware
//
#include <hip/hip_runtime.h>
#include <hip/hip_bf16.h>

// ---------------------------------------------------------------------------
// CDNA5 (gfx1250) WMMA bf16 path: V_WMMA_F32_16X16X32_BF16, wave32.
// DPP butterfly reductions (no LDS), async global->LDS K-tile staging.
// ---------------------------------------------------------------------------

typedef __attribute__((ext_vector_type(16))) __bf16 v16bf;
typedef __attribute__((ext_vector_type(8)))  float  v8f;
typedef __attribute__((ext_vector_type(4)))  int    v4i;

#define AS1 __attribute__((address_space(1)))
#define AS3 __attribute__((address_space(3)))

#if __has_builtin(__builtin_amdgcn_global_load_async_to_lds_b128)
#define HAVE_ASYNC_LDS 1
#else
#define HAVE_ASYNC_LDS 0
#endif

union Frag16 {
    v16bf    v;
    unsigned u[8];
    __bf16   e[16];
};

__device__ __forceinline__ v8f wmma_bf16(const Frag16& a, const Frag16& b, v8f c) {
    // (neg_a, A, neg_b, B, c_mod, C, reuse_a, reuse_b)
    return __builtin_amdgcn_wmma_f32_16x16x32_bf16(
        false, a.v, false, b.v, (short)0, c, false, false);
}

// A-matrix 16x32 (bf16): VGPR v holds bf16 pair (k,k+1); half selects +8/+24 band.
__device__ __forceinline__ int a_kbase(int v, int half) {
    return ((v >= 4) ? 16 : 0) + half * 8 + 2 * (v & 3);
}
// B-matrix 32x16 (bf16): same K order as A; VGPR v holds pair (k,k+1); half selects +8/+24 band.
__device__ __forceinline__ int b_kbase(int v, int half) {
    return ((v >= 4) ? 16 : 0) + half * 8 + 2 * (v & 3);
}

// ---- DPP 16-lane butterfly reductions (pure VALU, all lanes get result) ----
template <int CTRL>
__device__ __forceinline__ float dpp_xchg(float x) {
    int r = __builtin_amdgcn_update_dpp(0, __builtin_bit_cast(int, x),
                                        CTRL, 0xF, 0xF, true);
    return __builtin_bit_cast(float, r);
}
__device__ __forceinline__ float dpp_max16(float x) {
    x = fmaxf(x, dpp_xchg<0xB1>(x));    // quad_perm [1,0,3,2]  (xor 1)
    x = fmaxf(x, dpp_xchg<0x4E>(x));    // quad_perm [2,3,0,1]  (xor 2)
    x = fmaxf(x, dpp_xchg<0x141>(x));   // row_half_mirror      (quads in 8)
    x = fmaxf(x, dpp_xchg<0x140>(x));   // row_mirror           (8s in 16)
    return x;
}
__device__ __forceinline__ float dpp_add16(float x) {
    x += dpp_xchg<0xB1>(x);
    x += dpp_xchg<0x4E>(x);
    x += dpp_xchg<0x141>(x);
    x += dpp_xchg<0x140>(x);
    return x;
}

// ---- async global->LDS copy of 16B (bf16 x8), with sync fallback ----------
__device__ __forceinline__ void copy16_to_lds(const __bf16* gsrc, __bf16* ldst) {
#if HAVE_ASYNC_LDS
    __builtin_amdgcn_global_load_async_to_lds_b128(
        (AS1 v4i*)gsrc, (AS3 v4i*)ldst, 0, 0);
#else
#pragma unroll
    for (int j = 0; j < 4; ++j)
        ((unsigned*)ldst)[j] = ((const unsigned*)gsrc)[j];
#endif
}
__device__ __forceinline__ void wait_async_lds() {
#if HAVE_ASYNC_LDS
#if __has_builtin(__builtin_amdgcn_s_wait_asynccnt)
    __builtin_amdgcn_s_wait_asynccnt(0);
#else
    asm volatile("s_wait_asynccnt 0" ::: "memory");
#endif
#endif
}

// ---------------------------------------------------------------------------
// Generic  Y[M,N] = X[M,K] * W[N,K]^T + bias[N]
// 256 threads (8 waves), tile 128(M) x 64(N), K stepped by 32.
// ---------------------------------------------------------------------------
template <typename OutT>
__global__ void __launch_bounds__(256)
gemm_xwt(const float* __restrict__ X, const float* __restrict__ W,
         const float* __restrict__ bias, OutT* __restrict__ Y,
         int M, int N, int K) {
    __shared__ __bf16 Xs[128][34];
    __shared__ __bf16 Ws[64][34];

    const int tid  = threadIdx.x;
    const int lane = tid & 31;
    const int wave = tid >> 5;
    const int half = lane >> 4;
    const int l16  = lane & 15;
    const int wm   = wave >> 1;
    const int wn   = wave & 1;
    const int rowBlock = blockIdx.y * 128;
    const int colBlock = blockIdx.x * 64;

    v8f acc[2][2] = {};

    const int xRow = tid >> 1;
    const int xCol = (tid & 1) * 16;
    const int wRow = tid >> 2;
    const int wCol = (tid & 3) * 8;

    const float* xPtr = X + (size_t)(rowBlock + xRow) * K + xCol;
    const float* wPtr = W + (size_t)(colBlock + wRow) * K + wCol;

    for (int k0 = 0; k0 < K; k0 += 32) {
        // prefetch next K-tile into caches (global_prefetch_b8)
        if (k0 + 32 < K) {
            __builtin_prefetch(xPtr + k0 + 32, 0, 1);
            __builtin_prefetch(wPtr + k0 + 32, 0, 1);
        }
        {
            const float* src = xPtr + k0;
#pragma unroll
            for (int j = 0; j < 4; ++j) {
                float4 f = *(const float4*)(src + 4 * j);
                __bf16* dst = &Xs[xRow][xCol + 4 * j];
                dst[0] = (__bf16)f.x; dst[1] = (__bf16)f.y;
                dst[2] = (__bf16)f.z; dst[3] = (__bf16)f.w;
            }
            const float* srw = wPtr + k0;
#pragma unroll
            for (int j = 0; j < 2; ++j) {
                float4 f = *(const float4*)(srw + 4 * j);
                __bf16* dst = &Ws[wRow][wCol + 4 * j];
                dst[0] = (__bf16)f.x; dst[1] = (__bf16)f.y;
                dst[2] = (__bf16)f.z; dst[3] = (__bf16)f.w;
            }
        }
        __syncthreads();

        Frag16 a[2], b[2];
#pragma unroll
        for (int i = 0; i < 2; ++i) {
            const int r = wm * 32 + i * 16 + l16;
#pragma unroll
            for (int v = 0; v < 8; ++v)
                a[i].u[v] = *(const unsigned*)&Xs[r][a_kbase(v, half)];
        }
#pragma unroll
        for (int j = 0; j < 2; ++j) {
            const int c = wn * 32 + j * 16 + l16;
#pragma unroll
            for (int v = 0; v < 8; ++v)
                b[j].u[v] = *(const unsigned*)&Ws[c][b_kbase(v, half)];
        }
#pragma unroll
        for (int i = 0; i < 2; ++i)
#pragma unroll
            for (int j = 0; j < 2; ++j)
                acc[i][j] = wmma_bf16(a[i], b[j], acc[i][j]);
        __syncthreads();
    }

#pragma unroll
    for (int i = 0; i < 2; ++i) {
#pragma unroll
        for (int j = 0; j < 2; ++j) {
            const int col = colBlock + wn * 32 + j * 16 + l16;
            const float bv = bias[col];
#pragma unroll
            for (int v = 0; v < 8; ++v) {
                const int row = rowBlock + wm * 32 + i * 16 + v + 8 * half;
                Y[(size_t)row * N + col] = (OutT)(acc[i][j][v] + bv);
            }
        }
    }
}

// ---------------------------------------------------------------------------
// Flash attention on bf16 Q,K,V [8192,512]; row=(s*4+b), col=h*64+d.
// Block = 8 waves (one head); wave owns 16 q-rows; 32-key steps, double-
// buffered K/V tiles in LDS; K staged with GLOBAL_LOAD_ASYNC_TO_LDS_B128.
// Per step per wave: 4 WMMA scores + 4 WMMA O+=P*V.
// ---------------------------------------------------------------------------
__global__ void __launch_bounds__(256)
attn_fwd(const __bf16* __restrict__ Qb, const __bf16* __restrict__ Kb,
         const __bf16* __restrict__ Vb, float* __restrict__ Ob) {
    constexpr int S  = 2048;
    constexpr int LD = 512;

    __shared__ __align__(16) __bf16 Ks [2][32][72]; // [buf][key][d], 16B-aligned rows
    __shared__ __bf16 Vst[2][64][34];               // [buf][d][key] (transposed)
    __shared__ __bf16 Pt [8][16][34];               // per-wave P re-layout

    const int tid  = threadIdx.x;
    const int lane = tid & 31;
    const int wave = tid >> 5;
    const int half = lane >> 4;
    const int l16  = lane & 15;

    const int bh      = blockIdx.y;
    const int b       = bh >> 3;
    const int h       = bh & 7;
    const int colbase = h * 64;
    const int qtile   = blockIdx.x * 8 + wave;

    // Preload Q A-fragments (two 16x32 chunks over head_dim=64)
    Frag16 aq[2];
    {
        const int s = qtile * 16 + l16;
        const __bf16* qrow = Qb + (size_t)(s * 4 + b) * LD + colbase;
#pragma unroll
        for (int c = 0; c < 2; ++c)
#pragma unroll
            for (int v = 0; v < 8; ++v)
                aq[c].u[v] = *(const unsigned*)(qrow + 32 * c + a_kbase(v, half));
    }

    v8f   acc[4] = {};
    float mrow[8], lrow[8];
#pragma unroll
    for (int v = 0; v < 8; ++v) { mrow[v] = -1.0e30f; lrow[v] = 0.0f; }

    const int ck = tid >> 3;          // key 0..31
    const int cd = (tid & 7) * 8;     // d base (16B granule)

    auto load_tile = [&](int kb, int buf) {
        const __bf16* krow = Kb + (size_t)((kb + ck) * 4 + b) * LD + colbase + cd;
        copy16_to_lds(krow, &Ks[buf][ck][cd]);          // async -> ASYNCcnt
        const __bf16* vrow = Vb + (size_t)((kb + ck) * 4 + b) * LD + colbase + cd;
#pragma unroll
        for (int j = 0; j < 8; ++j)                     // transposed, ds_store_b16
            Vst[buf][cd + j][ck] = vrow[j];
    };

    int buf = 0;
    load_tile(0, 0);
    wait_async_lds();
    __syncthreads();

    for (int kb = 0; kb < S; kb += 32) {
        if (kb + 32 < S) load_tile(kb + 32, buf ^ 1);   // overlap with compute

        // S = Q * K^T : two 16x16 tiles (keys 0-15, 16-31)
        v8f sc[2] = {};
#pragma unroll
        for (int t = 0; t < 2; ++t) {
#pragma unroll
            for (int c = 0; c < 2; ++c) {
                Frag16 bk;
#pragma unroll
                for (int v = 0; v < 8; ++v)
                    bk.u[v] = *(const unsigned*)&Ks[buf][t * 16 + l16][32 * c + b_kbase(v, half)];
                sc[t] = wmma_bf16(aq[c], bk, sc[t]);
            }
        }

        // Online softmax; row m = v + 8*half lives across this half's 16 lanes.
        float p0[8], p1[8];
#pragma unroll
        for (int v = 0; v < 8; ++v) {
            float s0 = sc[0][v] * 0.125f;    // 1/sqrt(64)
            float s1 = sc[1][v] * 0.125f;
            float rmax = dpp_max16(fmaxf(s0, s1));
            const float mnew  = fmaxf(mrow[v], rmax);
            const float alpha = __expf(mrow[v] - mnew);
            mrow[v] = mnew;
            p0[v] = __expf(s0 - mnew);
            p1[v] = __expf(s1 - mnew);
            lrow[v] = lrow[v] * alpha + dpp_add16(p0[v] + p1[v]);
#pragma unroll
            for (int t = 0; t < 4; ++t) acc[t][v] *= alpha;
        }

        // Re-layout P (D layout -> A fragment) through wave-private LDS.
#pragma unroll
        for (int v = 0; v < 8; ++v) {
            const int m = v + 8 * half;
            Pt[wave][m][l16]      = (__bf16)p0[v];
            Pt[wave][m][16 + l16] = (__bf16)p1[v];
        }
        __builtin_amdgcn_wave_barrier();
        asm volatile("s_wait_dscnt 0" ::: "memory");    // DS pipe in-order per wave

        Frag16 pa;
#pragma unroll
        for (int v = 0; v < 8; ++v)
            pa.u[v] = *(const unsigned*)&Pt[wave][l16][a_kbase(v, half)];

        // O += P * V  (4 d-tiles, 32 keys per WMMA)
#pragma unroll
        for (int t = 0; t < 4; ++t) {
            Frag16 bv;
#pragma unroll
            for (int v = 0; v < 8; ++v)
                bv.u[v] = *(const unsigned*)&Vst[buf][t * 16 + l16][b_kbase(v, half)];
            acc[t] = wmma_bf16(pa, bv, acc[t]);
        }

        wait_async_lds();
        __syncthreads();
        buf ^= 1;
    }

    // Normalize and write f32 attention output [8192,512]
#pragma unroll
    for (int t = 0; t < 4; ++t) {
#pragma unroll
        for (int v = 0; v < 8; ++v) {
            const int m   = v + 8 * half;
            const int s   = qtile * 16 + m;
            const int col = colbase + t * 16 + l16;
            Ob[(size_t)(s * 4 + b) * LD + col] = acc[t][v] / lrow[v];
        }
    }
}

// ---------------------------------------------------------------------------
extern "C" void kernel_launch(void* const* d_in, const int* in_sizes, int n_in,
                              void* d_out, int out_size, void* d_ws, size_t ws_size,
                              hipStream_t stream) {
    const float* query = (const float*)d_in[0];
    const float* key   = (const float*)d_in[1];
    const float* value = (const float*)d_in[2];
    const float* w_in  = (const float*)d_in[3];   // [1536,1024]
    const float* b_in  = (const float*)d_in[4];   // [1536]
    const float* w_out = (const float*)d_in[5];   // [1024,512]
    const float* b_out = (const float*)d_in[6];   // [1024]
    float*       out   = (float*)d_out;           // [2048,4,1024]

    const int M  = 8192;   // S*B
    const int E  = 1024;
    const int BN = 512;    // bottleneck

    // Workspace: Q,K,V bf16 (3 x 8 MB) + attn-out f32 (16 MB) = 40 MB
    char*   ws = (char*)d_ws;
    __bf16* Qb = (__bf16*)(ws);
    __bf16* Kb = (__bf16*)(ws + (size_t)M * BN * 2);
    __bf16* Vb = (__bf16*)(ws + (size_t)M * BN * 4);
    float*  At = (float*) (ws + (size_t)M * BN * 6);

    dim3 blk(256);

    dim3 g1(BN / 64, M / 128);
    gemm_xwt<__bf16><<<g1, blk, 0, stream>>>(query, w_in,                      b_in,          Qb, M, BN, E);
    gemm_xwt<__bf16><<<g1, blk, 0, stream>>>(key,   w_in + (size_t)BN * E,     b_in + BN,     Kb, M, BN, E);
    gemm_xwt<__bf16><<<g1, blk, 0, stream>>>(value, w_in + (size_t)2 * BN * E, b_in + 2 * BN, Vb, M, BN, E);

    dim3 ga(16, 32);
    attn_fwd<<<ga, blk, 0, stream>>>(Qb, Kb, Vb, At);

    dim3 g2(E / 64, M / 128);
    gemm_xwt<float><<<g2, blk, 0, stream>>>(At, w_out, b_out, out, M, E, BN);
}
